// ActionRelationEncoder_15822659518548
// MI455X (gfx1250) — matrix operand order repaired and checked by reading, not compiled, not run on hardware
//
#include <hip/hip_runtime.h>
#include <hip/hip_bf16.h>

typedef __attribute__((ext_vector_type(16))) __bf16 v16bf;
typedef __attribute__((ext_vector_type(8)))  float  v8f;
typedef unsigned int u32x4 __attribute__((ext_vector_type(4)));
typedef int          i32x4 __attribute__((ext_vector_type(4)));
typedef int          i32x8 __attribute__((ext_vector_type(8)));

static constexpr int cB  = 64;
static constexpr int cN  = 128;
static constexpr int cNG = 64;
static constexpr int cH  = 16;
static constexpr int cVD = 2048;
static constexpr int cQD = 1024;
static constexpr int cOD = 1024;
static constexpr int cPD = 64;
static constexpr int cDG = 64;   // OD / H

enum { GF_RELU = 1, GF_OUT_F32 = 2, GF_OUT_B16 = 8, GF_B16_TPOSE = 16 };

union FragU { uint4 u[2]; v16bf v; };

#if __has_builtin(__builtin_amdgcn_tensor_load_to_lds)
#define USE_TDM 1
#else
#define USE_TDM 0
#endif

// LDS B-tile staging geometry: 32 rows x 64 bf16 data + 4-dword pad per row
// row stride = 32 + 4 dwords = 144 bytes = 72 bf16 elements (16B aligned)
static constexpr int LROW  = 72;              // elements per LDS row
static constexpr int LBUF  = 32 * LROW;       // elements per buffer (2304)
static constexpr int LWAVE = 8 * LROW;        // elements per wave slice (576)

// ---------------------------------------------------------------------------
// Elementwise converts
// ---------------------------------------------------------------------------
__global__ void to_bf16_kernel(const float* __restrict__ x, __bf16* __restrict__ y, int n) {
  for (int i = blockIdx.x * blockDim.x + threadIdx.x; i < n; i += gridDim.x * blockDim.x)
    y[i] = (__bf16)x[i];
}

// W (Nrows x Kcols) f32 row-major  ->  Wt (Kcols x Nrows) bf16 row-major
__global__ void transpose_bf16_kernel(const float* __restrict__ Wm, __bf16* __restrict__ Wt,
                                      int Nrows, int Kcols) {
  int total = Nrows * Kcols;
  for (int i = blockIdx.x * blockDim.x + threadIdx.x; i < total; i += gridDim.x * blockDim.x) {
    int k = i / Nrows;
    int n = i - k * Nrows;
    Wt[i] = (__bf16)Wm[(size_t)n * Kcols + k];
  }
}

// ---------------------------------------------------------------------------
// logpos[d][b,h,n,m] = log(max(pos_emb[b,n,m,:]*Wp[d,h,:] + bp[d,h], EPS)) + (Wb+bb)
// layout: ((b*16+h)*128+n)*64+m   (matches aff layout)
// ---------------------------------------------------------------------------
__global__ void pos_kernel(const float* __restrict__ pe, const float* __restrict__ Wp,
                           const float* __restrict__ bp, const float* __restrict__ Wb,
                           const float* __restrict__ bb,
                           float* __restrict__ lp0, float* __restrict__ lp1) {
  __shared__ float sW[2 * 16 * 64];
  __shared__ float sb[32];
  for (int i = threadIdx.x; i < 2048; i += blockDim.x) sW[i] = Wp[i];
  if (threadIdx.x < 32) sb[threadIdx.x] = bp[threadIdx.x];
  __syncthreads();
  float biasScalar = Wb[0] + bb[0];

  size_t gid = (size_t)blockIdx.x * blockDim.x + threadIdx.x;  // one per (b,n,m)
  const float* pr = pe + gid * cPD;
  float r[64];
#pragma unroll
  for (int i = 0; i < 16; ++i) {
    float4 t4 = reinterpret_cast<const float4*>(pr)[i];
    r[4 * i + 0] = t4.x; r[4 * i + 1] = t4.y; r[4 * i + 2] = t4.z; r[4 * i + 3] = t4.w;
  }
  int b = (int)(gid >> 13);
  int n = (int)((gid >> 6) & 127);
  int m = (int)(gid & 63);
  for (int d = 0; d < 2; ++d) {
    float* lp = d ? lp1 : lp0;
    for (int h = 0; h < 16; ++h) {
      float s = sb[d * 16 + h];
      const float* wrow = &sW[(d * 16 + h) * 64];
#pragma unroll
      for (int p = 0; p < 64; ++p) s = fmaf(r[p], wrow[p], s);
      s = fmaxf(s, 1e-6f);  // max(relu(x), eps) == max(x, eps)
      lp[((size_t)(b * 16 + h) * cN + n) * cNG + m] = __logf(s) + biasScalar;
    }
  }
}

// ---------------------------------------------------------------------------
// mask + concat: vcq[row, 0:1024] = bf16(act); vcq[row, 1024:2048] = sum!=0 ? q[b] : 0
// ---------------------------------------------------------------------------
__global__ void mask_concat_kernel(const float* __restrict__ act, const __bf16* __restrict__ qb,
                                   __bf16* __restrict__ vcq) {
  __shared__ float red[256];
  int row = blockIdx.x;
  int b = row >> 7;
  const float* ar = act + (size_t)row * cOD;
  float s = 0.f;
  for (int i = threadIdx.x; i < cOD; i += 256) s += ar[i];
  red[threadIdx.x] = s;
  __syncthreads();
  for (int o = 128; o > 0; o >>= 1) {
    if ((int)threadIdx.x < o) red[threadIdx.x] += red[threadIdx.x + o];
    __syncthreads();
  }
  bool nz = red[0] != 0.f;
  __bf16* outr = vcq + (size_t)row * (cOD + cQD);
  const __bf16* qr = qb + (size_t)b * cQD;
  for (int i = threadIdx.x; i < cOD; i += 256) {
    outr[i] = (__bf16)ar[i];
    outr[cOD + i] = nz ? qr[i] : (__bf16)0.f;
  }
}

// ---------------------------------------------------------------------------
// TDM issue: DMA an 8-row x 64-col bf16 slice of Bt into LDS (padded rows).
// All operands are scalar (SGPR). Each wave moves its own slice.
// D# group0: {count=1, lds_addr, global_addr[56:0], type=2}
// D# group1: data_size=2B, pad_enable, pad_interval=32dw, pad_amount=4dw,
//            tile_dim0=64, tile_dim1=8, tensor dims huge (no OOB), stride0=ldb
// ---------------------------------------------------------------------------
#if USE_TDM
__device__ inline void tdm_issue_b_slice(const __bf16* gsrc, unsigned ldsOff, int ldb) {
  uint64_t ga = (uint64_t)(uintptr_t)gsrc;
  u32x4 g0 = { 1u, ldsOff, (unsigned)ga,
               (unsigned)((ga >> 32) & 0x1FFFFFFull) | (2u << 30) };
  i32x8 g1 = { (int)0x07110000,      // data_size=2B | pad_enable | interval=32dw | amount=4dw
               0,                    // atomic barrier addr = 0
               0x4000,               // tensor_dim0 = 0x40000000 (hi16)
               (64 << 16) | 0x4000,  // tile_dim0=64 | tensor_dim1 hi16
               8,                    // tile_dim1 = 8, tile_dim2 = 0
               ldb,                  // tensor_dim0_stride lo32
               0, 0 };
  i32x4 z4 = {};
#if defined(__clang_major__) && (__clang_major__ >= 23)
  i32x8 z8 = {};
  __builtin_amdgcn_tensor_load_to_lds(g0, g1, z4, z4, z8, 0);
#else
  __builtin_amdgcn_tensor_load_to_lds(g0, g1, z4, z4, 0);
#endif
}
#endif

// ---------------------------------------------------------------------------
// Generic WMMA GEMM: C[M,Ntot] = A[M,K](bf16,row-major) x Bt[K,Ntot](bf16) + bias
// Block = 128 threads (4 waves); block tile 64x64; wave tile 16x64.
// B tile staged in LDS by the Tensor Data Mover, double-buffered on TENSORcnt.
// Optional A row remap (take/skip) to pull kv rows out of self_feat.
// ---------------------------------------------------------------------------
__global__ void gemm_wmma_kernel(const __bf16* __restrict__ A, int lda, int aTake, int aSkip,
                                 const __bf16* __restrict__ Bt, int ldb,
                                 const float* __restrict__ bias,
                                 float* __restrict__ outF32, __bf16* __restrict__ outB16,
                                 int Ntot, int K, int tposeGroup, int flags) {
  __shared__ __align__(16) __bf16 sB[2 * LBUF];

  int lane = threadIdx.x & 31;
  int waveId = __builtin_amdgcn_readfirstlane((int)(threadIdx.x >> 5));  // scalar
  int mlane = lane & 15;
  int half = lane >> 4;
  int mBlock = blockIdx.x * 64 + waveId * 16;
  int nBlock = blockIdx.y * 64;

  int lrow = mBlock + mlane;
  int prow = (aTake > 0) ? (lrow / aTake) * aSkip + (lrow % aTake) : lrow;
  const __bf16* Arow = A + (size_t)prow * lda;

  v8f acc[4] = {};

#if USE_TDM
  unsigned ldsBase = (unsigned)(uintptr_t)(&sB[0]);
  const __bf16* gB0 = Bt + nBlock;
  // prologue: stage K-step 0 into buffer 0 (each wave moves 8 rows)
  tdm_issue_b_slice(gB0 + (size_t)(waveId * 8) * ldb,
                    ldsBase + (unsigned)(waveId * LWAVE) * 2u, ldb);
  int cur = 0;
  for (int kk = 0; kk < K; kk += 32) {
    __builtin_amdgcn_s_wait_tensorcnt(0);
    __syncthreads();
    if (kk + 32 < K)
      tdm_issue_b_slice(gB0 + (size_t)(kk + 32 + waveId * 8) * ldb,
                        ldsBase + (unsigned)((cur ^ 1) * LBUF + waveId * LWAVE) * 2u, ldb);
    FragU fa;
    const __bf16* pa = Arow + kk + half * 8;
    fa.u[0] = *reinterpret_cast<const uint4*>(pa);
    fa.u[1] = *reinterpret_cast<const uint4*>(pa + 16);
    const __bf16* pb = sB + cur * LBUF + mlane;
#pragma unroll
    for (int t = 0; t < 4; ++t) {
      FragU fb;
#pragma unroll
      for (int s = 0; s < 16; ++s) fb.v[s] = pb[(8 * half + (s & 7) + ((s >> 3) << 4)) * LROW + t * 16];
      acc[t] = __builtin_amdgcn_wmma_f32_16x16x32_bf16(false, fa.v, false, fb.v,
                                                       (short)0, acc[t], false, false);
    }
    cur ^= 1;
  }
#else
  for (int kk = 0; kk < K; kk += 32) {
    FragU fa;
    const __bf16* pa = Arow + kk + half * 8;
    fa.u[0] = *reinterpret_cast<const uint4*>(pa);
    fa.u[1] = *reinterpret_cast<const uint4*>(pa + 16);
    const __bf16* pb = Bt + (size_t)kk * ldb + nBlock + mlane;
#pragma unroll
    for (int t = 0; t < 4; ++t) {
      FragU fb;
#pragma unroll
      for (int s = 0; s < 16; ++s) fb.v[s] = pb[(size_t)(8 * half + (s & 7) + ((s >> 3) << 4)) * ldb + t * 16];
      acc[t] = __builtin_amdgcn_wmma_f32_16x16x32_bf16(false, fa.v, false, fb.v,
                                                       (short)0, acc[t], false, false);
    }
  }
#endif

#pragma unroll
  for (int t = 0; t < 4; ++t) {
#pragma unroll
    for (int r = 0; r < 8; ++r) {
      int rowl = mBlock + 8 * half + r;
      int col = nBlock + t * 16 + mlane;
      float vv = acc[t][r];
      if (bias) vv += bias[col];
      if (flags & GF_RELU) vv = fmaxf(vv, 0.f);
      if (flags & GF_OUT_F32) outF32[(size_t)rowl * Ntot + col] = vv;
      if (flags & GF_OUT_B16) {
        __bf16 bvv = (__bf16)vv;
        if (flags & GF_B16_TPOSE)
          outB16[((size_t)(rowl / tposeGroup) * Ntot + col) * tposeGroup + (rowl % tposeGroup)] = bvv;
        else
          outB16[(size_t)rowl * Ntot + col] = bvv;
      }
    }
  }
}

// ---------------------------------------------------------------------------
// aff[b,h,n,m] = (qh[b,n,h,:]·kh[b,m,h,:]) / 8 + logpos[b,h,n,m]
// grid = B*H blocks, 256 threads (8 waves: n-tiles). K = DG = 64.
// ---------------------------------------------------------------------------
__global__ void aff_kernel(const __bf16* __restrict__ qh, const __bf16* __restrict__ khT,
                           const float* __restrict__ lp, float* __restrict__ aff) {
  int lane = threadIdx.x & 31, wave = threadIdx.x >> 5;
  int mlane = lane & 15, half = lane >> 4;
  int bh = blockIdx.x, b = bh >> 4, h = bh & 15;
  int nrow = wave * 16;
  const __bf16* Arow = qh + ((size_t)(b * cN + nrow + mlane)) * cOD + h * cDG;
  const __bf16* Bbase = khT + (size_t)b * (cOD * cNG) + (size_t)(h * cDG) * cNG;
  v8f acc[4] = {};
#pragma unroll
  for (int kk = 0; kk < 64; kk += 32) {
    FragU fa;
    const __bf16* pa = Arow + kk + half * 8;
    fa.u[0] = *reinterpret_cast<const uint4*>(pa);
    fa.u[1] = *reinterpret_cast<const uint4*>(pa + 16);
    const __bf16* pb = Bbase + (size_t)kk * cNG + mlane;
#pragma unroll
    for (int t = 0; t < 4; ++t) {
      FragU fb;
#pragma unroll
      for (int s = 0; s < 16; ++s) fb.v[s] = pb[(size_t)(8 * half + (s & 7) + ((s >> 3) << 4)) * cNG + t * 16];
      acc[t] = __builtin_amdgcn_wmma_f32_16x16x32_bf16(false, fa.v, false, fb.v,
                                                       (short)0, acc[t], false, false);
    }
  }
  size_t base = (size_t)bh * cN * cNG;
#pragma unroll
  for (int t = 0; t < 4; ++t) {
#pragma unroll
    for (int r = 0; r < 8; ++r) {
      int n_ = nrow + 8 * half + r, m_ = t * 16 + mlane;
      size_t idx = base + (size_t)n_ * cNG + m_;
      aff[idx] = acc[t][r] * 0.125f + lp[idx];
    }
  }
}

// ---------------------------------------------------------------------------
// softmax over m (64) per (b,h,n) row; output bf16 in same layout
// ---------------------------------------------------------------------------
__global__ void softmax_kernel(const float* __restrict__ aff, __bf16* __restrict__ att) {
  int lane = threadIdx.x & 31, wave = threadIdx.x >> 5;
  size_t row = (size_t)blockIdx.x * 8 + wave;
  const float* ar = aff + row * cNG;
  float a0 = ar[lane], a1 = ar[lane + 32];
  float mx = fmaxf(a0, a1);
#pragma unroll
  for (int o = 16; o > 0; o >>= 1) mx = fmaxf(mx, __shfl_xor(mx, o, 32));
  float e0 = __expf(a0 - mx), e1 = __expf(a1 - mx);
  float s = e0 + e1;
#pragma unroll
  for (int o = 16; o > 0; o >>= 1) s += __shfl_xor(s, o, 32);
  float inv = 1.f / s;
  __bf16* orow = att + row * cNG;
  orow[lane] = (__bf16)(e0 * inv);
  orow[lane + 32] = (__bf16)(e1 * inv);
}

// ---------------------------------------------------------------------------
// out_buf[b,n,h*64+g] += att[b,h,n,:] · KW[b,:,h*64+g] + bout[h*64+g]
// grid = B*H blocks, 256 threads. K = NG = 64.
// ---------------------------------------------------------------------------
__global__ void attn_out_kernel(const __bf16* __restrict__ att, const __bf16* __restrict__ KW,
                                const float* __restrict__ boutd, float* __restrict__ outb) {
  int lane = threadIdx.x & 31, wave = threadIdx.x >> 5;
  int mlane = lane & 15, half = lane >> 4;
  int bh = blockIdx.x, b = bh >> 4, h = bh & 15;
  int nrow = wave * 16;
  const __bf16* Arow = att + ((size_t)bh * cN + nrow + mlane) * cNG;
  const __bf16* Bbase = KW + (size_t)b * cNG * cOD + h * cDG;
  v8f acc[4] = {};
#pragma unroll
  for (int kk = 0; kk < 64; kk += 32) {
    FragU fa;
    const __bf16* pa = Arow + kk + half * 8;
    fa.u[0] = *reinterpret_cast<const uint4*>(pa);
    fa.u[1] = *reinterpret_cast<const uint4*>(pa + 16);
    const __bf16* pb = Bbase + (size_t)kk * cOD + mlane;
#pragma unroll
    for (int t = 0; t < 4; ++t) {
      FragU fb;
#pragma unroll
      for (int s = 0; s < 16; ++s) fb.v[s] = pb[(size_t)(8 * half + (s & 7) + ((s >> 3) << 4)) * cOD + t * 16];
      acc[t] = __builtin_amdgcn_wmma_f32_16x16x32_bf16(false, fa.v, false, fb.v,
                                                       (short)0, acc[t], false, false);
    }
  }
#pragma unroll
  for (int t = 0; t < 4; ++t) {
#pragma unroll
    for (int r = 0; r < 8; ++r) {
      int n_ = nrow + 8 * half + r, g_ = t * 16 + mlane;
      size_t idx = (size_t)(b * cN + n_) * cOD + h * cDG + g_;
      outb[idx] += acc[t][r] + boutd[h * cDG + g_];
    }
  }
}

// act += relu(out_buf)
__global__ void add_relu_kernel(float* __restrict__ act, const float* __restrict__ outb, int n) {
  for (int i = blockIdx.x * blockDim.x + threadIdx.x; i < n; i += gridDim.x * blockDim.x)
    act[i] += fmaxf(outb[i], 0.f);
}

// ---------------------------------------------------------------------------
extern "C" void kernel_launch(void* const* d_in, const int* in_sizes, int n_in,
                              void* d_out, int out_size, void* d_ws, size_t ws_size,
                              hipStream_t stream) {
  (void)in_sizes; (void)n_in; (void)out_size; (void)ws_size;
  const float* v    = (const float*)d_in[0];
  const float* pe   = (const float*)d_in[1];
  const float* q    = (const float*)d_in[2];
  const float* Wv   = (const float*)d_in[3];
  const float* bv   = (const float*)d_in[4];
  const float* Ws   = (const float*)d_in[5];
  const float* bs   = (const float*)d_in[6];
  const float* Wb   = (const float*)d_in[7];
  const float* bb   = (const float*)d_in[8];
  const float* Wq   = (const float*)d_in[9];
  const float* bq   = (const float*)d_in[10];
  const float* Wk   = (const float*)d_in[11];
  const float* bk   = (const float*)d_in[12];
  const float* Wp   = (const float*)d_in[13];
  const float* bp   = (const float*)d_in[14];
  const float* Wout = (const float*)d_in[15];
  const float* bout = (const float*)d_in[16];
  float* act = (float*)d_out;  // act_v lives in d_out across steps

  const int M  = cB * cN;   // 8192
  const int Mk = cB * cNG;  // 4096

  char* wsp = (char*)d_ws;
  auto alloc = [&](size_t bytes) -> void* {
    void* p = (void*)wsp;
    wsp += (bytes + 255) & ~(size_t)255;
    return p;
  };
  __bf16* v_b   = (__bf16*)alloc((size_t)M * cVD * 2);
  __bf16* q_b   = (__bf16*)alloc((size_t)cB * cQD * 2);
  __bf16* Wvt   = (__bf16*)alloc((size_t)cVD * cOD * 2);
  __bf16* Wst   = (__bf16*)alloc((size_t)(cOD + cQD) * cOD * 2);
  __bf16* Wqt   = (__bf16*)alloc((size_t)2 * cOD * cOD * 2);
  __bf16* Wkt   = (__bf16*)alloc((size_t)2 * cOD * cOD * 2);
  __bf16* Wot   = (__bf16*)alloc((size_t)2 * cOD * cOD * 2);
  float*  lp0   = (float*)alloc((size_t)cB * cH * cN * cNG * 4);
  float*  lp1   = (float*)alloc((size_t)cB * cH * cN * cNG * 4);
  __bf16* vcq   = (__bf16*)alloc((size_t)M * (cOD + cQD) * 2);
  float*  outb  = (float*)alloc((size_t)M * cOD * 4);
  __bf16* selfb = (__bf16*)alloc((size_t)M * cOD * 2);
  __bf16* qh_b  = (__bf16*)alloc((size_t)M * cOD * 2);
  __bf16* khT   = (__bf16*)alloc((size_t)Mk * cOD * 2);
  __bf16* KW    = (__bf16*)alloc((size_t)Mk * cOD * 2);
  float*  aff   = (float*)alloc((size_t)cB * cH * cN * cNG * 4);
  __bf16* att   = (__bf16*)alloc((size_t)cB * cH * cN * cNG * 2);

  // One-time converts / transposes / position log-weights
  to_bf16_kernel<<<4096, 256, 0, stream>>>(v, v_b, M * cVD);
  to_bf16_kernel<<<64, 256, 0, stream>>>(q, q_b, cB * cQD);
  transpose_bf16_kernel<<<4096, 256, 0, stream>>>(Wv, Wvt, cOD, cVD);
  transpose_bf16_kernel<<<4096, 256, 0, stream>>>(Ws, Wst, cOD, cOD + cQD);
  for (int d = 0; d < 2; ++d) {
    size_t off = (size_t)d * cOD * cOD;
    transpose_bf16_kernel<<<2048, 256, 0, stream>>>(Wq + off, Wqt + off, cOD, cOD);
    transpose_bf16_kernel<<<2048, 256, 0, stream>>>(Wk + off, Wkt + off, cOD, cOD);
    transpose_bf16_kernel<<<2048, 256, 0, stream>>>(Wout + off, Wot + off, cOD, cOD);
  }
  pos_kernel<<<(cB * cN * cNG) / 128, 128, 0, stream>>>(pe, Wp, bp, Wb, bb, lp0, lp1);

  // act = relu(v @ Wv^T + bv)
  gemm_wmma_kernel<<<dim3(M / 64, cOD / 64), 128, 0, stream>>>(
      v_b, cVD, 0, 0, Wvt, cOD, bv, act, nullptr, cOD, cVD, 0, GF_RELU | GF_OUT_F32);

  for (int step = 0; step < 2; ++step) {
    mask_concat_kernel<<<M, 256, 0, stream>>>(act, q_b, vcq);
    // self_feat -> outb (f32 accumulator) + selfb (bf16 operand)
    gemm_wmma_kernel<<<dim3(M / 64, cOD / 64), 128, 0, stream>>>(
        vcq, cOD + cQD, 0, 0, Wst, cOD, bs, outb, selfb, cOD, cOD + cQD, 0,
        GF_OUT_F32 | GF_OUT_B16);
    for (int d = 0; d < 2; ++d) {
      size_t off = (size_t)d * cOD * cOD;
      // qh (bf16, natural layout)
      gemm_wmma_kernel<<<dim3(M / 64, cOD / 64), 128, 0, stream>>>(
          selfb, cOD, 0, 0, Wqt + off, cOD, bq + d * cOD, nullptr, qh_b, cOD, cOD, 0,
          GF_OUT_B16);
      // kh, written transposed per-batch: khT[b][c][m]
      gemm_wmma_kernel<<<dim3(Mk / 64, cOD / 64), 128, 0, stream>>>(
          selfb, cOD, cNG, cN, Wkt + off, cOD, bk + d * cOD, nullptr, khT, cOD, cOD, cNG,
          GF_OUT_B16 | GF_B16_TPOSE);
      // KW[b,m,hg] = kv[b,m,:] @ Wout[d]^T (natural layout, no bias)
      gemm_wmma_kernel<<<dim3(Mk / 64, cOD / 64), 128, 0, stream>>>(
          selfb, cOD, cNG, cN, Wot + off, cOD, nullptr, nullptr, KW, cOD, cOD, 0,
          GF_OUT_B16);
      aff_kernel<<<cB * cH, 256, 0, stream>>>(qh_b, khT, d ? lp1 : lp0, aff);
      softmax_kernel<<<(cB * cH * cN) / 8, 256, 0, stream>>>(aff, att);
      attn_out_kernel<<<cB * cH, 256, 0, stream>>>(att, KW, bout + d * cOD, outb);
    }
    add_relu_kernel<<<4096, 256, 0, stream>>>(act, outb, M * cOD);
  }
}
